// ConfGNN_42253888258098
// MI455X (gfx1250) — compile-verified, matrix operand order checked by reading, not run on hardware
//
#include <hip/hip_runtime.h>
#include <hip/hip_bf16.h>

// ---------------------------------------------------------------------------
// Problem constants (match reference)
// ---------------------------------------------------------------------------
#define DLAT   256
#define HID    512
#define STEPS  12
#define GRAPHS 64
#define NPG    64
#define EPG    1024
#define NNODES (GRAPHS * NPG)   // 4096
#define NEDGES (GRAPHS * EPG)   // 65536

typedef __bf16 bf16;
typedef bf16 v16bf __attribute__((ext_vector_type(16)));
typedef bf16 v8bf  __attribute__((ext_vector_type(8)));
typedef bf16 v4bf  __attribute__((ext_vector_type(4)));
typedef float v8f  __attribute__((ext_vector_type(8)));

union FragAB { v16bf v; v8bf h[2]; };

// ---------------------------------------------------------------------------
// Tensor Data Mover support (guarded; falls back to sync LDS staging)
// ---------------------------------------------------------------------------
#if defined(__HIP_DEVICE_COMPILE__) && \
    __has_builtin(__builtin_amdgcn_tensor_load_to_lds) && \
    __has_builtin(__builtin_amdgcn_s_wait_tensorcnt)
#define TDM_OK 1
#else
#define TDM_OK 0
#endif

#if TDM_OK
typedef unsigned int v4u __attribute__((ext_vector_type(4)));
typedef int v8i __attribute__((ext_vector_type(8)));
typedef int v4i __attribute__((ext_vector_type(4)));

__device__ inline unsigned lds_off(const void* p)
{
    // generic -> LDS address space; AS3 pointers are the raw LDS byte offset
    return (unsigned)(unsigned long long)
        (__attribute__((address_space(3))) const char*)p;
}

// 2D tile load: tile [tile1 rows x tile0 cols] of bf16 from a row-major
// tensor with row stride `stride0` elements; rows beyond dim1_rem zero-fill.
__device__ inline void tdm_load_tile_bf16(unsigned lds_addr, const bf16* gptr,
                                          unsigned dim0_rem, unsigned dim1_rem,
                                          unsigned tile0, unsigned tile1,
                                          unsigned stride0)
{
    unsigned long long ga = (unsigned long long)gptr;
    v4u g0;
    g0[0] = 1u;                                   // count=1, user descriptor
    g0[1] = lds_addr;                             // LDS byte address
    g0[2] = (unsigned)(ga & 0xffffffffu);         // global addr lo
    g0[3] = (unsigned)((ga >> 32) & 0x01ffffffu)  // global addr hi (57b)
            | (2u << 30);                         // type = 2 ("image")
    v8i g1;
    g1[0] = (int)(1u << 16);                      // wg_mask=0, data_size=1 (2B)
    g1[1] = (int)((dim0_rem & 0xffffu) << 16);    // tensor_dim0 lo16
    g1[2] = (int)((dim0_rem >> 16) | ((dim1_rem & 0xffffu) << 16));
    g1[3] = (int)((dim1_rem >> 16) | (tile0 << 16));   // tile_dim0
    g1[4] = (int)tile1;                           // tile_dim1, tile_dim2=0
    g1[5] = (int)stride0;                         // tensor_dim0_stride lo32
    g1[6] = 0;                                    // stride hi, dim1_stride lo
    g1[7] = 0;
    v4i z4 = {0, 0, 0, 0};
    v8i z8 = {0, 0, 0, 0, 0, 0, 0, 0};
    // 6-arg form (clang-23 / therock-10.0 headers): g0, g1, g2, g3, g4, cpol
    __builtin_amdgcn_tensor_load_to_lds(g0, g1, z4, z4, z8, 0);
}
#endif

// ---------------------------------------------------------------------------
// BF16 WMMA GEMM:  C[M,N] = act( A[M,K](bf16) @ Bt[N,K](bf16, pre-transposed)
//                               + bias )
//   RELU && !RESIDUAL : Cout(bf16) = relu(acc + bias)
//   RESIDUAL          : Cres(f32) += acc + bias
// Block: 256 threads (8 waves), tile 128(M) x 128(N), K-step 32.
// Wave grid 4(M) x 2(N); each wave: 2x4 tiles of v_wmma_f32_16x16x32_bf16.
// Double-buffered LDS; next K-tile fetched by the Tensor Data Mover while
// the current tile's WMMAs execute (TDM overlap), sync fallback otherwise.
// ---------------------------------------------------------------------------
template <bool RELU, bool RESIDUAL>
__global__ __launch_bounds__(256)
void gemm_bf16_wmma(const bf16* __restrict__ A, const bf16* __restrict__ Bt,
                    const float* __restrict__ bias,
                    float* __restrict__ Cres, bf16* __restrict__ Cout,
                    int M, int N, int K)
{
    constexpr int BM = 128, BN = 128, BK = 32;
    __shared__ bf16 As[2][BM * BK];   // 2 x 8 KB, packed row-major [m][k]
    __shared__ bf16 Bs[2][BN * BK];   // 2 x 8 KB, packed row-major [n][k]

    const int tid    = threadIdx.x;
    const int lane   = tid & 31;
    const int wave   = tid >> 5;
    const int waveM  = (wave >> 1) * 32;   // 0,32,64,96
    const int waveN  = (wave & 1) * 64;    // 0,64
    const int blockM = blockIdx.y * BM;
    const int blockN = blockIdx.x * BN;

    v8f acc[2][4] = {};

    // one K-step of WMMAs from LDS buffer `buf`
    auto mma_step = [&](int buf) {
        const int halfk = (lane >> 4) * 8;
        FragAB a[2], b[4];
#pragma unroll
        for (int mt = 0; mt < 2; ++mt) {
            int r = waveM + mt * 16 + (lane & 15);
            a[mt].h[0] = *(const v8bf*)&As[buf][r * BK + halfk];
            a[mt].h[1] = *(const v8bf*)&As[buf][r * BK + halfk + 16];
        }
#pragma unroll
        for (int nt = 0; nt < 4; ++nt) {
            int c = waveN + nt * 16 + (lane & 15);
            b[nt].h[0] = *(const v8bf*)&Bs[buf][c * BK + halfk];
            b[nt].h[1] = *(const v8bf*)&Bs[buf][c * BK + halfk + 16];
        }
#pragma unroll
        for (int mt = 0; mt < 2; ++mt)
#pragma unroll
            for (int nt = 0; nt < 4; ++nt)
                acc[mt][nt] = __builtin_amdgcn_wmma_f32_16x16x32_bf16(
                    false, a[mt].v, false, b[nt].v, (short)0, acc[mt][nt],
                    false, false);
    };

#if TDM_OK
    auto issue = [&](int buf, int k0) {
        tdm_load_tile_bf16(lds_off(As[buf]), A + (size_t)blockM * K + k0,
                           (unsigned)(K - k0), (unsigned)(M - blockM),
                           BK, BM, (unsigned)K);
        tdm_load_tile_bf16(lds_off(Bs[buf]), Bt + (size_t)blockN * K + k0,
                           (unsigned)(K - k0), (unsigned)(N - blockN),
                           BK, BN, (unsigned)K);
    };
    // prologue: fetch first K-tile
    if (wave == 0) {
        issue(0, 0);
        __builtin_amdgcn_s_wait_tensorcnt(0);
    }
    __syncthreads();
    int cur = 0;
    for (int k0 = 0; k0 < K; k0 += BK) {
        const bool more = (k0 + BK) < K;
        if (wave == 0 && more) issue(cur ^ 1, k0 + BK);  // overlap DMA w/ math
        mma_step(cur);
        if (wave == 0 && more) __builtin_amdgcn_s_wait_tensorcnt(0);
        __syncthreads();   // all LDS reads of `cur` done before it is reused
        cur ^= 1;
    }
#else
    for (int k0 = 0; k0 < K; k0 += BK) {
        {   // ---- stage A tile: 128 rows x 32 bf16; 2 threads/row, 16 each
            int r  = tid >> 1;
            int c  = (tid & 1) * 16;
            int gm = blockM + r;
            uint4 p0 = {0, 0, 0, 0}, p1 = {0, 0, 0, 0};
            if (gm < M) {
                const uint4* s = (const uint4*)(A + (size_t)gm * K + k0 + c);
                p0 = s[0];
                p1 = s[1];
            }
            *(uint4*)&As[0][r * BK + c]     = p0;
            *(uint4*)&As[0][r * BK + c + 8] = p1;
        }
        {   // ---- stage B tile from pre-transposed bf16 weights
            int r = tid >> 1;
            int c = (tid & 1) * 16;
            const uint4* s = (const uint4*)(Bt + (size_t)(blockN + r) * K + k0 + c);
            uint4 p0 = s[0];
            uint4 p1 = s[1];
            *(uint4*)&Bs[0][r * BK + c]     = p0;
            *(uint4*)&Bs[0][r * BK + c + 8] = p1;
        }
        __syncthreads();
        mma_step(0);
        __syncthreads();
    }
#endif

    // ---- epilogue: C layout: vgpr v, lanes 0-15 -> (M=v, N=lane); 16-31 -> M=v+8
#pragma unroll
    for (int mt = 0; mt < 2; ++mt) {
#pragma unroll
        for (int nt = 0; nt < 4; ++nt) {
            int gn  = blockN + waveN + nt * 16 + (lane & 15);
            int gm0 = blockM + waveM + mt * 16 + (lane >> 4) * 8;
            float bv = bias[gn];
#pragma unroll
            for (int v = 0; v < 8; ++v) {
                int gm = gm0 + v;
                if (gm < M) {
                    float r = acc[mt][nt][v] + bv;
                    if (RESIDUAL) {
                        Cres[(size_t)gm * N + gn] += r;
                    } else {
                        if (RELU) r = r > 0.f ? r : 0.f;
                        Cout[(size_t)gm * N + gn] = (bf16)r;
                    }
                }
            }
        }
    }
}

// ---------------------------------------------------------------------------
// Weight pre-pass: out[n*K + k] = bf16(in[k*N + n])  (transpose + convert)
// ---------------------------------------------------------------------------
__global__ __launch_bounds__(256)
void wconv_kernel(const float* __restrict__ in, bf16* __restrict__ out,
                  int K, int N)
{
    size_t idx = (size_t)blockIdx.x * blockDim.x + threadIdx.x;
    if (idx >= (size_t)K * N) return;
    int n = (int)(idx / K);
    int k = (int)(idx % K);
    out[idx] = (bf16)in[(size_t)k * N + n];
}

// ---------------------------------------------------------------------------
// Gather/concat kernels (f32 sources -> packed bf16 rows)
// ---------------------------------------------------------------------------
__global__ __launch_bounds__(256)
void edge_concat_kernel(const float* __restrict__ x, const float* __restrict__ e,
                        const float* __restrict__ u, const int* __restrict__ src,
                        const int* __restrict__ dst, const int* __restrict__ eb,
                        bf16* __restrict__ ein)
{
    size_t idx = ((size_t)blockIdx.x * blockDim.x + threadIdx.x) * 4; // E*1024
    int edge = (int)(idx >> 10);
    int col  = (int)(idx & 1023);
    const float* s;
    if (col < 256)      s = x + (size_t)src[edge] * DLAT + col;
    else if (col < 512) s = x + (size_t)dst[edge] * DLAT + (col - 256);
    else if (col < 768) s = e + (size_t)edge * DLAT + (col - 512);
    else                s = u + (size_t)eb[edge] * DLAT + (col - 768);
    float4 f = *(const float4*)s;
    v4bf o = {(bf16)f.x, (bf16)f.y, (bf16)f.z, (bf16)f.w};
    *(v4bf*)(ein + idx) = o;
}

__global__ __launch_bounds__(256)
void node_concat_kernel(const float* __restrict__ x, const float* __restrict__ ssent,
                        const float* __restrict__ srecv, const float* __restrict__ u,
                        const int* __restrict__ nb, bf16* __restrict__ nin)
{
    size_t idx = ((size_t)blockIdx.x * blockDim.x + threadIdx.x) * 4; // N*1024
    int node = (int)(idx >> 10);
    int col  = (int)(idx & 1023);
    const float* s;
    if (col < 256)      s = x + (size_t)node * DLAT + col;
    else if (col < 512) s = ssent + (size_t)node * DLAT + (col - 256);
    else if (col < 768) s = srecv + (size_t)node * DLAT + (col - 512);
    else                s = u + (size_t)nb[node] * DLAT + (col - 768);
    float4 f = *(const float4*)s;
    v4bf o = {(bf16)f.x, (bf16)f.y, (bf16)f.z, (bf16)f.w};
    *(v4bf*)(nin + idx) = o;
}

__global__ __launch_bounds__(256)
void g_concat_kernel(const float* __restrict__ gx, const float* __restrict__ ge,
                     const float* __restrict__ u, bf16* __restrict__ gin)
{
    size_t idx = ((size_t)blockIdx.x * blockDim.x + threadIdx.x) * 4; // G*768
    if (idx >= (size_t)GRAPHS * 768) return;
    int g   = (int)(idx / 768);
    int col = (int)(idx % 768);
    const float* s;
    if (col < 256)      s = gx + (size_t)g * DLAT + col;
    else if (col < 512) s = ge + (size_t)g * DLAT + (col - 256);
    else                s = u + (size_t)g * DLAT + (col - 512);
    float4 f = *(const float4*)s;
    v4bf o = {(bf16)f.x, (bf16)f.y, (bf16)f.z, (bf16)f.w};
    *(v4bf*)(gin + idx) = o;
}

__global__ __launch_bounds__(256)
void f32_to_bf16_kernel(const float* __restrict__ in, bf16* __restrict__ out,
                        size_t n4)
{
    size_t idx = (size_t)blockIdx.x * blockDim.x + threadIdx.x;
    if (idx >= n4) return;
    float4 f = *(const float4*)(in + idx * 4);
    v4bf o = {(bf16)f.x, (bf16)f.y, (bf16)f.z, (bf16)f.w};
    *(v4bf*)(out + idx * 4) = o;
}

// ---------------------------------------------------------------------------
// Per-graph segment sum of edge features into nodes (LDS float atomics).
// grid.x = graph, grid.y = feature chunk (0/1, 128 feats each). 256 threads.
// ---------------------------------------------------------------------------
__global__ __launch_bounds__(256)
void seg_sum_kernel(const float* __restrict__ e, const int* __restrict__ idx,
                    float* __restrict__ out)
{
    __shared__ float acc[NPG * 128];   // 32 KB
    const int g = blockIdx.x, chunk = blockIdx.y, tid = threadIdx.x;
    for (int i = tid; i < NPG * 128; i += 256) acc[i] = 0.f;
    __syncthreads();
    const int f = (tid & 127) + chunk * 128;
    for (int eo = (tid >> 7); eo < EPG; eo += 2) {
        int ge    = g * EPG + eo;
        int local = idx[ge] - g * NPG;
        float v   = e[(size_t)ge * DLAT + f];
        atomicAdd(&acc[local * 128 + (tid & 127)], v);
    }
    __syncthreads();
    for (int i = tid; i < NPG * 128; i += 256) {
        int node = i >> 7, ff = i & 127;
        out[(size_t)(g * NPG + node) * DLAT + chunk * 128 + ff] = acc[i];
    }
}

// Sum 64 consecutive rows per graph: out[g,f] = sum_i in[g*64+i, f]
__global__ __launch_bounds__(256)
void graph_sum_kernel(const float* __restrict__ in, float* __restrict__ out)
{
    int g = blockIdx.x, f = threadIdx.x;
    float s = 0.f;
#pragma unroll 4
    for (int i = 0; i < NPG; ++i) s += in[(size_t)(g * NPG + i) * DLAT + f];
    out[(size_t)g * DLAT + f] = s;
}

// pos head layer 2: pos[n,j] = sum_k ph[n,k]*W[k,j] + b[j]   (3 outputs)
__global__ __launch_bounds__(256)
void pos_out_kernel(const bf16* __restrict__ ph, const float* __restrict__ W,
                    const float* __restrict__ b, float* __restrict__ pos)
{
    int i = blockIdx.x * blockDim.x + threadIdx.x;
    if (i >= NNODES * 3) return;
    int n = i / 3, j = i % 3;
    float s = b[j];
#pragma unroll 8
    for (int k = 0; k < DLAT; ++k) s += (float)ph[(size_t)n * DLAT + k] * W[k * 3 + j];
    pos[i] = s;
}

// per-graph mean centering of pos; block per graph, 192 threads = 64 nodes x 3
__global__ __launch_bounds__(192)
void center_kernel(float* __restrict__ pos)
{
    __shared__ float s[3];
    int g = blockIdx.x, t = threadIdx.x;
    if (t < 3) s[t] = 0.f;
    __syncthreads();
    int node = t / 3, j = t % 3;
    float v = pos[(size_t)(g * NPG + node) * 3 + j];
    atomicAdd(&s[j], v);
    __syncthreads();
    pos[(size_t)(g * NPG + node) * 3 + j] = v - s[j] * (1.0f / NPG);
}

// ---------------------------------------------------------------------------
// Host orchestration
// ---------------------------------------------------------------------------
extern "C" void kernel_launch(void* const* d_in, const int* in_sizes, int n_in,
                              void* d_out, int out_size, void* d_ws, size_t ws_size,
                              hipStream_t stream)
{
    (void)in_sizes; (void)n_in; (void)out_size; (void)ws_size;

    const float* x0 = (const float*)d_in[0];
    const float* e0 = (const float*)d_in[1];
    const float* u0 = (const float*)d_in[2];
    const int* edge_index = (const int*)d_in[3];
    const int* node_batch = (const int*)d_in[4];
    const int* edge_batch = (const int*)d_in[5];
    const int* srcI = edge_index;
    const int* dstI = edge_index + NEDGES;

    const float* P[22];
    for (int i = 0; i < 22; ++i) P[i] = (const float*)d_in[6 + i];
    // param order: eW1 eb1 eW2 eb2 eW3 eb3 | nW1 nb1 nW2 nb2 nW3 nb3 |
    //              gW1 gb1 gW2 gb2 gW3 gb3 | pW1 pb1 pW2 pb2
    const size_t st[22] = {
        4ull*DLAT*HID, HID, (size_t)HID*HID, HID, (size_t)HID*DLAT, DLAT,
        4ull*DLAT*HID, HID, (size_t)HID*HID, HID, (size_t)HID*DLAT, DLAT,
        3ull*DLAT*HID, HID, (size_t)HID*HID, HID, (size_t)HID*DLAT, DLAT,
        (size_t)DLAT*DLAT, DLAT, (size_t)DLAT*3, 3
    };

    // ---- workspace carve-up
    char* w = (char*)d_ws;
    auto alloc = [&](size_t bytes) -> void* {
        void* p = (void*)w;
        w += (bytes + 255) & ~(size_t)255;
        return p;
    };
    float* e     = (float*)alloc((size_t)NEDGES * DLAT * 4);
    float* xb    = (float*)alloc((size_t)NNODES * DLAT * 4);
    float* ub    = (float*)alloc((size_t)GRAPHS * DLAT * 4);
    bf16*  ein   = (bf16*) alloc((size_t)NEDGES * 4 * DLAT * 2);
    bf16*  h1    = (bf16*) alloc((size_t)NEDGES * HID * 2);
    bf16*  h2    = (bf16*) alloc((size_t)NEDGES * HID * 2);
    float* ssent = (float*)alloc((size_t)NNODES * DLAT * 4);
    float* srecv = (float*)alloc((size_t)NNODES * DLAT * 4);
    float* gx    = (float*)alloc((size_t)GRAPHS * DLAT * 4);
    float* ge    = (float*)alloc((size_t)GRAPHS * DLAT * 4);
    bf16*  nin   = (bf16*) alloc((size_t)NNODES * 4 * DLAT * 2);
    bf16*  gin   = (bf16*) alloc((size_t)GRAPHS * 3 * DLAT * 2);
    bf16*  gh1   = (bf16*) alloc((size_t)GRAPHS * HID * 2);
    bf16*  gh2   = (bf16*) alloc((size_t)GRAPHS * HID * 2);
    bf16*  xbf   = (bf16*) alloc((size_t)NNODES * DLAT * 2);
    bf16*  ph    = (bf16*) alloc((size_t)NNODES * DLAT * 2);
    float* pos   = (float*)d_out;

    // ---- bf16 transposed weights (Wt[N][K]) for all steps
    struct WSpec { int pidx; int K; int N; };
    const WSpec wspec[10] = {
        {0, 4 * DLAT, HID}, {2, HID, HID}, {4, HID, DLAT},     // edge MLP
        {6, 4 * DLAT, HID}, {8, HID, HID}, {10, HID, DLAT},    // node MLP
        {12, 3 * DLAT, HID}, {14, HID, HID}, {16, HID, DLAT},  // global MLP
        {18, DLAT, DLAT}                                        // pos layer 1
    };
    size_t wtper = 0;
    for (int m = 0; m < 10; ++m) wtper += (size_t)wspec[m].K * wspec[m].N;
    bf16* wtbase = (bf16*)alloc((size_t)STEPS * wtper * 2);
    bf16* wt[STEPS][10];
    {
        size_t off = 0;
        for (int p = 0; p < STEPS; ++p)
            for (int m = 0; m < 10; ++m) {
                wt[p][m] = wtbase + off;
                off += (size_t)wspec[m].K * wspec[m].N;
            }
    }
    for (int p = 0; p < STEPS; ++p)
        for (int m = 0; m < 10; ++m) {
            const WSpec& s = wspec[m];
            size_t n = (size_t)s.K * s.N;
            wconv_kernel<<<(n + 255) / 256, 256, 0, stream>>>(
                P[s.pidx] + (size_t)p * st[s.pidx], wt[p][m], s.K, s.N);
        }

    // ---- initial state copies (d2d, graph-capturable)
    (void)hipMemcpyAsync(e,  e0, (size_t)NEDGES * DLAT * 4, hipMemcpyDeviceToDevice, stream);
    (void)hipMemcpyAsync(xb, x0, (size_t)NNODES * DLAT * 4, hipMemcpyDeviceToDevice, stream);
    (void)hipMemcpyAsync(ub, u0, (size_t)GRAPHS * DLAT * 4, hipMemcpyDeviceToDevice, stream);

    auto gemm_relu = [&](const bf16* A, const bf16* Bt, const float* bias,
                         bf16* out, int M, int N, int K) {
        dim3 grid((N + 127) / 128, (M + 127) / 128);
        gemm_bf16_wmma<true, false><<<grid, dim3(256), 0, stream>>>(
            A, Bt, bias, (float*)nullptr, out, M, N, K);
    };
    auto gemm_res = [&](const bf16* A, const bf16* Bt, const float* bias,
                        float* out, int M, int N, int K) {
        dim3 grid((N + 127) / 128, (M + 127) / 128);
        gemm_bf16_wmma<false, true><<<grid, dim3(256), 0, stream>>>(
            A, Bt, bias, out, (bf16*)nullptr, M, N, K);
    };

    for (int p = 0; p < STEPS; ++p) {
        const float* eb1 = P[1]  + p * st[1];
        const float* eb2 = P[3]  + p * st[3];
        const float* eb3 = P[5]  + p * st[5];
        const float* nb1 = P[7]  + p * st[7];
        const float* nb2 = P[9]  + p * st[9];
        const float* nb3 = P[11] + p * st[11];
        const float* gb1 = P[13] + p * st[13];
        const float* gb2 = P[15] + p * st[15];
        const float* gb3 = P[17] + p * st[17];
        const float* pb1 = P[19] + p * st[19];
        const float* pW2 = P[20] + p * st[20];
        const float* pb2 = P[21] + p * st[21];

        // --- edge model: e += MLP(concat(x[src], x[dst], e, u[eb]))
        edge_concat_kernel<<<(size_t)NEDGES * 1024 / 4 / 256, 256, 0, stream>>>(
            xb, e, ub, srcI, dstI, edge_batch, ein);
        gemm_relu(ein, wt[p][0], eb1, h1, NEDGES, HID, 4 * DLAT);
        gemm_relu(h1,  wt[p][1], eb2, h2, NEDGES, HID, HID);
        gemm_res (h2,  wt[p][2], eb3, e,  NEDGES, DLAT, HID);

        // --- segment sums of new e by src / dst
        seg_sum_kernel<<<dim3(GRAPHS, 2), 256, 0, stream>>>(e, srcI, ssent);
        seg_sum_kernel<<<dim3(GRAPHS, 2), 256, 0, stream>>>(e, dstI, srecv);

        // --- node model: x += MLP(concat(x, ssent, srecv, u[nb]))
        node_concat_kernel<<<(size_t)NNODES * 1024 / 4 / 256, 256, 0, stream>>>(
            xb, ssent, srecv, ub, node_batch, nin);
        gemm_relu(nin, wt[p][3], nb1, h1, NNODES, HID, 4 * DLAT);
        gemm_relu(h1,  wt[p][4], nb2, h2, NNODES, HID, HID);
        gemm_res (h2,  wt[p][5], nb3, xb, NNODES, DLAT, HID);

        // --- global model: u += MLP(concat(sum_nodes(x), sum_edges(e), u))
        graph_sum_kernel<<<GRAPHS, DLAT, 0, stream>>>(xb, gx);
        graph_sum_kernel<<<GRAPHS, DLAT, 0, stream>>>(ssent, ge);
        g_concat_kernel<<<(GRAPHS * 768 / 4 + 255) / 256, 256, 0, stream>>>(
            gx, ge, ub, gin);
        gemm_relu(gin, wt[p][6], gb1, gh1, GRAPHS, HID, 3 * DLAT);
        gemm_relu(gh1, wt[p][7], gb2, gh2, GRAPHS, HID, HID);
        gemm_res (gh2, wt[p][8], gb3, ub,  GRAPHS, DLAT, HID);

        // --- position head (only last step's output is returned)
        if (p == STEPS - 1) {
            f32_to_bf16_kernel<<<((size_t)NNODES * DLAT / 4 + 255) / 256, 256, 0,
                                 stream>>>(xb, xbf, (size_t)NNODES * DLAT / 4);
            gemm_relu(xbf, wt[p][9], pb1, ph, NNODES, DLAT, DLAT);
            pos_out_kernel<<<(NNODES * 3 + 255) / 256, 256, 0, stream>>>(
                ph, pW2, pb2, pos);
            center_kernel<<<GRAPHS, 192, 0, stream>>>(pos);
        }
    }
}
